// SparseRollingCorrelationGraph_19842748907724
// MI455X (gfx1250) — matrix operand order repaired and checked by reading, not compiled, not run on hardware
//
#include <hip/hip_runtime.h>
#include <hip/hip_bf16.h>

typedef __attribute__((ext_vector_type(16))) _Float16 v16h;
typedef __attribute__((ext_vector_type(8)))  _Float16 v8h;
typedef __attribute__((ext_vector_type(8)))  float    v8f;

#define EPS_F 1e-6f
#define TOPK  20
#define LFIX  256              // history length (reference setup)
#define KP    (LFIX + 8)       // padded LDS row stride in f16: 528B = 16B-aligned,
                               // rotates banks row-to-row -> conflict-free frag reads

// ---------------------------------------------------------------------------
// Kernel 1: per-row center + variance-normalize (z-score), apply row mask,
// emit f16 copy for the WMMA GEMM.  One 256-thread block per (b,n) row.
// ---------------------------------------------------------------------------
__global__ void corr_normalize_f16(const float* __restrict__ hist,
                                   const unsigned char* __restrict__ mask,
                                   _Float16* __restrict__ xn,
                                   int L)
{
    const int row = blockIdx.x;           // 0 .. B*N-1
    const int tid = threadIdx.x;          // 0 .. 255  (== L)
    __shared__ float red[256];

    float x = hist[(size_t)row * L + tid];

    red[tid] = x;
    __syncthreads();
    for (int s = 128; s > 0; s >>= 1) {
        if (tid < s) red[tid] += red[tid + s];
        __syncthreads();
    }
    const float mean = red[0] / (float)L;
    __syncthreads();

    const float c = x - mean;
    red[tid] = c * c;
    __syncthreads();
    for (int s = 128; s > 0; s >>= 1) {
        if (tid < s) red[tid] += red[tid + s];
        __syncthreads();
    }
    const float denom = fmaxf(sqrtf(red[0] / (float)L), EPS_F);
    const float m = mask[row] ? 1.0f : 0.0f;

    xn[(size_t)row * L + tid] = (_Float16)((c / denom) * m);
}

// ---------------------------------------------------------------------------
// Kernel 2: sim = Z * Z^T / L via v_wmma_f32_16x16x32_f16 with async
// global->LDS tile staging.  Block = 8 waves = 64(M) x 32(N) output tile:
//   - async-copy A strip (64 rows x 256 K) and B strip (32 rows x 256 K)
//     into padded LDS (global_load_async_to_lds_b128, ASYNCcnt)
//   - each wave computes one 16x16 tile from LDS fragments, K unrolled.
// ---------------------------------------------------------------------------
__device__ __forceinline__ v16h make_frag(const _Float16* p0, const _Float16* p1)
{
    v8h lo = *(const v8h*)p0;             // 16-byte LDS load
    v8h hi = *(const v8h*)p1;             // 16-byte LDS load
    v16h r;
#pragma unroll
    for (int i = 0; i < 8; ++i) { r[i] = lo[i]; r[i + 8] = hi[i]; }
    return r;
}

__global__ __launch_bounds__(256)
void corr_sim_wmma(const _Float16* __restrict__ xn,
                   float* __restrict__ out,
                   int N)
{
    const int tid  = threadIdx.x;
    const int lane = tid & 31;
    const int w    = tid >> 5;                 // wave 0..7
    const int b      = blockIdx.z;
    const int mblock = blockIdx.y << 6;        // 64-row strip
    const int nblock = blockIdx.x << 5;        // 32-col strip

    __shared__ __align__(16) _Float16 sA[64 * KP];   // 33,792 B
    __shared__ __align__(16) _Float16 sB[32 * KP];   // 16,896 B

    const _Float16* Xb = xn + (size_t)b * N * LFIX;
    const unsigned sA_u = (unsigned)(size_t)(const void*)sA;   // LDS byte offset
    const unsigned sB_u = (unsigned)(size_t)(const void*)sB;

    // ---- async global -> LDS staging, 16B chunks (per-lane addresses) ----
    // A: 64 rows * 32 chunks = 2048; B: 32 rows * 32 chunks = 1024.
    for (int c = tid; c < 3072; c += 256) {    // uniform branch per iteration
        const int col = c & 31;                // 16B chunk within row
        const _Float16* g;
        unsigned l;
        if (c < 2048) {
            const int row = c >> 5;
            g = Xb + (size_t)(mblock + row) * LFIX + (col << 3);
            l = sA_u + ((unsigned)(row * KP + (col << 3)) << 1);
        } else {
            const int row = (c - 2048) >> 5;
            g = Xb + (size_t)(nblock + row) * LFIX + (col << 3);
            l = sB_u + ((unsigned)(row * KP + (col << 3)) << 1);
        }
        asm volatile("global_load_async_to_lds_b128 %0, %1, off"
                     :: "v"(l), "v"((unsigned long long)(size_t)g)
                     : "memory");
    }
    asm volatile("s_wait_asynccnt 0x0" ::: "memory");   // my wave's copies done
    __syncthreads();                                    // whole tile visible

    // ---- WMMA from LDS fragments ----
    const int half = lane >> 4;                // 0 | 1
    const int l16  = lane & 15;
    const int wm   = (w & 3) << 4;             // 0,16,32,48
    const int wn   = (w >> 2) << 4;            // 0,16
    const int akoff = half << 3;               // A-layout K sub-offset
    const int bkoff = half << 3;               // B-layout K sub-offset

    const _Float16* arow = sA + (wm + l16) * KP;   // A: lane -> row l&15
    const _Float16* brow = sB + (wn + l16) * KP;   // B: lane -> col l&15

    v8f acc = {};
#pragma unroll
    for (int ks = 0; ks < (LFIX >> 5); ++ks) {
        const int k0 = ks << 5;
        // A 16x32: elems 0..7 = K[kb..kb+8), 8..15 = K[kb+16..kb+24), kb=8*(l>>4)
        v16h a  = make_frag(arow + k0 + akoff, arow + k0 + akoff + 16);
        // B 32x16: elems 0..7 = K[kb..kb+8), 8..15 = K[kb+16..kb+24), kb=8*(l>>4)
        v16h bf = make_frag(brow + k0 + bkoff, brow + k0 + bkoff + 16);
        acc = __builtin_amdgcn_wmma_f32_16x16x32_f16(
                  false, a, false, bf, (short)0, acc, false, false);
    }

    const float invL = 1.0f / (float)LFIX;
    float* outB = out + (size_t)b * N * N;
#pragma unroll
    for (int r = 0; r < 8; ++r) {
        const int gm = mblock + wm + r + (half << 3);   // C/D layout
        const int gn = nblock + wn + l16;
        float v = acc[r] * invL;
        v = v > 0.0f ? v : 0.0f;                        // relu
        if (gm == gn) v = 0.0f;                         // zero diagonal
        outB[(size_t)gm * N + gn] = v;
    }
}

// ---------------------------------------------------------------------------
// Kernel 3: per-row top-20 + double row-normalization, in place on d_out.
// One 256-thread block per (b,n) row; row cached in 16KB LDS.
// ---------------------------------------------------------------------------
__global__ void topk_rownorm(float* __restrict__ out,
                             const unsigned char* __restrict__ mask,
                             int N)
{
    const int rid = blockIdx.x;                 // b*N + n
    const int b   = rid / N;
    const int n   = rid - b * N;
    const int tid = threadIdx.x;

    float* rowp = out + (size_t)b * N * N + (size_t)n * N;
    const unsigned char* mrow = mask + (size_t)b * N;

    __shared__ float svals[4096];
    __shared__ float rv[256];
    __shared__ int   ri[256];
    __shared__ int   sel_idx[TOPK];
    __shared__ float sel_val[TOPK];
    __shared__ float sscale;

    const float NEG_INF = -__builtin_inff();

    for (int i = tid; i < N; i += blockDim.x) {
        float v = rowp[i];
        if (!mrow[i]) v = NEG_INF;              // masked column -> -inf score
        svals[i] = v;
    }
    __syncthreads();

    for (int k = 0; k < TOPK; ++k) {
        float bv = NEG_INF;
        int   bi = 0x7fffffff;
        for (int i = tid; i < N; i += blockDim.x) {
            float v = svals[i];
            if (v > bv || (v == bv && i < bi)) { bv = v; bi = i; }
        }
        rv[tid] = bv; ri[tid] = bi;
        __syncthreads();
        for (int s = 128; s > 0; s >>= 1) {     // tie-break: lowest index
            if (tid < s) {
                float v2 = rv[tid + s]; int i2 = ri[tid + s];
                if (v2 > rv[tid] || (v2 == rv[tid] && i2 < ri[tid])) {
                    rv[tid] = v2; ri[tid] = i2;
                }
            }
            __syncthreads();
        }
        if (tid == 0) {
            const int   bi0 = ri[0];
            const float bv0 = rv[0];
            sel_idx[k] = bi0;
            sel_val[k] = (bv0 == NEG_INF) ? 0.0f : bv0;   // where(isfinite,v,0)
            svals[bi0] = NEG_INF;                          // remove from pool
        }
        __syncthreads();
    }

    if (tid == 0) {
        float s = 0.0f;
        for (int k = 0; k < TOPK; ++k) {
            float v = sel_val[k];
            if (sel_idx[k] == n) v = 0.0f;      // zero-diag before row sum
            s += v;
        }
        // _row_normalize applied twice collapses to one scale factor
        const float inv1 = 1.0f / fmaxf(s, EPS_F);
        const float s2   = s * inv1;
        const float inv2 = 1.0f / fmaxf(s2, EPS_F);
        sscale = inv1 * inv2;
    }
    __syncthreads();

    for (int i = tid; i < N; i += blockDim.x) rowp[i] = 0.0f;
    __syncthreads();                             // order zero-fill vs scatter
    if (tid < TOPK) {
        const int idx = sel_idx[tid];
        float v = sel_val[tid];
        if (idx == n) v = 0.0f;                  // zero diagonal
        rowp[idx] = v * sscale;                  // indices are distinct
    }
}

// ---------------------------------------------------------------------------
extern "C" void kernel_launch(void* const* d_in, const int* in_sizes, int n_in,
                              void* d_out, int out_size, void* d_ws, size_t ws_size,
                              hipStream_t stream)
{
    (void)n_in; (void)out_size; (void)ws_size;

    const float*         hist = (const float*)d_in[0];          // [B,N,L] f32
    const unsigned char* mask = (const unsigned char*)d_in[1];  // [B,N] bool
    float*               out  = (float*)d_out;                  // [B,N,N] f32

    const int BN = in_sizes[1];           // B*N = 16384
    const int L  = in_sizes[0] / BN;      // 256
    const int B  = 4;                     // per reference setup_inputs
    const int N  = BN / B;                // 4096

    _Float16* xn = (_Float16*)d_ws;       // f16 z-scores: B*N*L*2 = 8 MB

    // 1) z-score normalize rows -> f16
    corr_normalize_f16<<<BN, L, 0, stream>>>(hist, mask, xn, L);

    // 2) dense sim via WMMA with async LDS staging: 64x32 tile per block
    dim3 grid2(N >> 5, N >> 6, B);        // (128, 64, 4)
    corr_sim_wmma<<<grid2, 256, 0, stream>>>(xn, out, N);

    // 3) per-row top-20 + double row-normalize, in place
    topk_rownorm<<<BN, 256, 0, stream>>>(out, mask, N);
}
